// PAM_Module_10479720202889
// MI455X (gfx1250) — matrix operand order repaired and checked by reading, not compiled, not run on hardware
//
#include <hip/hip_runtime.h>

// PAM (position attention) module for MI455X / gfx1250, wave32 + WMMA f16.
// B=4, C=256, d=C/2=128, N=H*W=4096.

typedef __attribute__((ext_vector_type(16))) _Float16 v16h;
typedef __attribute__((ext_vector_type(8)))  _Float16 v8h;
typedef __attribute__((ext_vector_type(2)))  _Float16 v2h;
typedef __attribute__((ext_vector_type(8)))  float    v8f;
typedef __attribute__((ext_vector_type(4)))  float    fl4;
typedef __attribute__((ext_vector_type(4)))  int      v4i;

#define WMMA_F16(A, Bf, Cacc) \
  __builtin_amdgcn_wmma_f32_16x16x32_f16(false, (A), false, (Bf), (short)0, (Cacc), false, false)

// Optional CDNA5 async global->LDS copy path (guarded; falls back to v8h copies).
#if defined(__HIP_DEVICE_COMPILE__) && \
    __has_builtin(__builtin_amdgcn_global_load_async_to_lds_b128) && \
    __has_builtin(__builtin_amdgcn_s_wait_asynccnt)
#define HAVE_ASYNC_LDS 1
typedef __attribute__((address_space(1))) v4i gbl_v4i;
typedef __attribute__((address_space(3))) v4i lds_v4i;
#endif

static __device__ __forceinline__ v8f zero8() {
  v8f z = {0.f, 0.f, 0.f, 0.f, 0.f, 0.f, 0.f, 0.f};
  return z;
}

// A-matrix fragment (16x32 f16) from row-major LDS tile, leading dim = ld (f16 elems).
// CDNA5 layout: lanes 0-15 (row M=lane) hold K {0..7,16..23}; lanes 16-31 K {8..15,24..31}.
// Two contiguous 16B LDS loads per lane.
static __device__ __forceinline__ v16h frag_a(const _Float16* base, int ld, int lane) {
  const _Float16* p = base + (lane & 15) * ld + ((lane >> 4) << 3);
  v8h lo = *(const v8h*)(p);
  v8h hi = *(const v8h*)(p + 16);
  return __builtin_shufflevector(lo, hi, 0, 1, 2, 3, 4, 5, 6, 7, 8, 9, 10, 11, 12, 13, 14, 15);
}

// B-matrix fragment (32x16 f16) from a COLUMN-MAJOR LDS tile sBt[col][K] (ldk = K-chunk).
// Lanes 0-15 col=lane hold K {0..7,16..23}; lanes 16-31 hold K {8..15,24..31} -> two 16B loads/lane.
static __device__ __forceinline__ v16h frag_bt(const _Float16* base, int ldk, int lane) {
  const _Float16* p = base + (lane & 15) * ldk + ((lane >> 4) << 3);
  v8h lo = *(const v8h*)(p);
  v8h hi = *(const v8h*)(p + 16);
  return __builtin_shufflevector(lo, hi, 0, 1, 2, 3, 4, 5, 6, 7, 8, 9, 10, 11, 12, 13, 14, 15);
}

// ---------------------------------------------------------------------------
// Kernel 1: fused Q/K/V 1x1-conv projections (512x16384x256 GEMM).
// Rows 0..127 -> q (Qt[b][d][m], f16), 128..255 -> k (Kn[b][n][d], f16),
// 256..511 -> v (Vraw[b][c][n], f32, pre-softmax).  grid=(128,4), block=256.
// ---------------------------------------------------------------------------
__global__ __launch_bounds__(256) void pam_proj(
    const float* __restrict__ x,
    const float* __restrict__ Wq, const float* __restrict__ bq,
    const float* __restrict__ Wk, const float* __restrict__ bk,
    const float* __restrict__ Wv, const float* __restrict__ bv,
    _Float16* __restrict__ Qt, _Float16* __restrict__ Kn, float* __restrict__ Vraw) {
  __shared__ __align__(16) _Float16 sA[128 * 64];   // [row][K]   K-chunk = 64
  __shared__ __align__(16) _Float16 sBt[128 * 64];  // [col][K]
  const int tid  = threadIdx.x;
  const int lane = tid & 31;
  const int wave = tid >> 5;
  const int wrow = wave & 3;   // 4 row groups of 32
  const int wcol = wave >> 2;  // 2 col groups of 64
  const int rowTile = blockIdx.y;  // 0..3
  const int colTile = blockIdx.x;  // 0..127
  const int bb = colTile >> 5;
  const int n0 = (colTile & 31) * 128;

  const float* Wrow0;
  if (rowTile == 0)      Wrow0 = Wq;
  else if (rowTile == 1) Wrow0 = Wk;
  else                   Wrow0 = Wv + (size_t)(rowTile - 2) * 128 * 256;

  v8f acc[8];
#pragma unroll
  for (int i = 0; i < 8; ++i) acc[i] = zero8();

  for (int kc0 = 0; kc0 < 256; kc0 += 64) {
    __syncthreads();
    // A: weights f32 -> f16, row-major [128][64]
    for (int i = tid; i < 128 * 64; i += 256) {
      const int r = i >> 6, c = i & 63;
      sA[i] = (_Float16)Wrow0[r * 256 + kc0 + c];
    }
    // B: x f32 -> f16, transposed into [col][64]; 2 K-rows x 8 cols per chunk.
    for (int i = tid; i < 512; i += 256) {
      const int k2 = i >> 4, col8 = (i & 15) << 3;
      const int k = k2 << 1;
      const float* s0 = x + (size_t)(bb * 256 + kc0 + k) * 4096 + n0 + col8;
      fl4 r0a = *(const fl4*)(s0);
      fl4 r0b = *(const fl4*)(s0 + 4);
      fl4 r1a = *(const fl4*)(s0 + 4096);
      fl4 r1b = *(const fl4*)(s0 + 4100);
#pragma unroll
      for (int e = 0; e < 4; ++e) {
        v2h t0 = {(_Float16)r0a[e], (_Float16)r1a[e]};
        v2h t1 = {(_Float16)r0b[e], (_Float16)r1b[e]};
        *(v2h*)&sBt[(col8 + e) * 64 + k]     = t0;
        *(v2h*)&sBt[(col8 + 4 + e) * 64 + k] = t1;
      }
    }
    __syncthreads();
#pragma unroll
    for (int kk = 0; kk < 64; kk += 32) {
      const v16h a0 = frag_a(sA + (wrow * 32) * 64 + kk, 64, lane);
      const v16h a1 = frag_a(sA + (wrow * 32 + 16) * 64 + kk, 64, lane);
#pragma unroll
      for (int j = 0; j < 4; ++j) {
        const v16h bf = frag_bt(sBt + (wcol * 64 + j * 16) * 64 + kk, 64, lane);
        acc[j]     = WMMA_F16(a0, bf, acc[j]);
        acc[4 + j] = WMMA_F16(a1, bf, acc[4 + j]);
      }
    }
  }

  const int lc = lane & 15;
  const int lr = (lane >> 4) << 3;
#pragma unroll
  for (int ri = 0; ri < 2; ++ri) {
#pragma unroll
    for (int j = 0; j < 4; ++j) {
      const v8f c = acc[ri * 4 + j];
      const int colg = n0 + wcol * 64 + j * 16 + lc;
#pragma unroll
      for (int el = 0; el < 8; ++el) {
        const int rloc = wrow * 32 + ri * 16 + lr + el;  // 0..127 within tile
        float val = c[el];
        if (rowTile == 0) {
          val += bq[rloc];
          Qt[(size_t)(bb * 128 + rloc) * 4096 + colg] = (_Float16)val;
        } else if (rowTile == 1) {
          val += bk[rloc];
          Kn[(size_t)(bb * 4096 + colg) * 128 + rloc] = (_Float16)val;
        } else {
          const int cch = (rowTile - 2) * 128 + rloc;
          val += bv[cch];
          Vraw[(size_t)(bb * 256 + cch) * 4096 + colg] = val;
        }
      }
    }
  }
}

// ---------------------------------------------------------------------------
// Kernel 2: V spatial softmax: Vh[b][c][n] = softmax_n(Vraw[b][c][:]) as f16.
// ---------------------------------------------------------------------------
__global__ __launch_bounds__(256) void pam_vsoftmax(const float* __restrict__ Vraw,
                                                    _Float16* __restrict__ Vh) {
  __shared__ float red[256];
  const int t = threadIdx.x;
  const size_t row = blockIdx.x;
  const float* src = Vraw + row * 4096;
  float r[16];
  float mx = -3.4e38f;
#pragma unroll
  for (int i = 0; i < 16; ++i) { r[i] = src[t + i * 256]; mx = fmaxf(mx, r[i]); }
  red[t] = mx; __syncthreads();
  for (int s = 128; s > 0; s >>= 1) { if (t < s) red[t] = fmaxf(red[t], red[t + s]); __syncthreads(); }
  mx = red[0]; __syncthreads();
  float sm = 0.f;
#pragma unroll
  for (int i = 0; i < 16; ++i) { r[i] = __expf(r[i] - mx); sm += r[i]; }
  red[t] = sm; __syncthreads();
  for (int s = 128; s > 0; s >>= 1) { if (t < s) red[t] += red[t + s]; __syncthreads(); }
  const float rinv = 1.0f / red[0];
#pragma unroll
  for (int i = 0; i < 16; ++i) Vh[row * 4096 + t + i * 256] = (_Float16)(r[i] * rinv);
}

// ---------------------------------------------------------------------------
// Kernel 3: transposed energy e[n][m] = sum_d k[d,n]*q[m,d]  (one batch).
// A = Kn[b] (N x 128 row-major), B = Qt[b] (128 x N row-major). Out f32.
// grid = (32 mTiles, 32 nTiles), block 256.
// ---------------------------------------------------------------------------
__global__ __launch_bounds__(256) void pam_energy(const _Float16* __restrict__ Kn,
                                                  const _Float16* __restrict__ Qt,
                                                  float* __restrict__ Eng, int b) {
  __shared__ __align__(16) _Float16 sA[128 * 64];   // [n-row][K]
  __shared__ __align__(16) _Float16 sBt[128 * 64];  // [m-col][K]
  const int tid = threadIdx.x, lane = tid & 31, wave = tid >> 5;
  const int wrow = wave & 3, wcol = wave >> 2;
  const int m0 = blockIdx.x * 128, n0 = blockIdx.y * 128;

  v8f acc[8];
#pragma unroll
  for (int i = 0; i < 8; ++i) acc[i] = zero8();

  for (int kc0 = 0; kc0 < 128; kc0 += 64) {
    __syncthreads();
    // A panel: f16 -> f16 straight copy, 16B vectors (4 chunks/thread).
    for (int i = tid; i < 1024; i += 256) {
      const int r = i >> 3, c8 = (i & 7) << 3;
      *(v8h*)&sA[r * 64 + c8] =
          *(const v8h*)&Kn[(size_t)(b * 4096 + n0 + r) * 128 + kc0 + c8];
    }
    // B panel: coalesced row reads, transposed pair-stores (b32) into [col][64].
    for (int i = tid; i < 512; i += 256) {
      const int k2 = i >> 4, col8 = (i & 15) << 3;
      const int k = k2 << 1;
      const _Float16* s0 = Qt + (size_t)(b * 128 + kc0 + k) * 4096 + m0 + col8;
      v8h q0 = *(const v8h*)(s0);
      v8h q1 = *(const v8h*)(s0 + 4096);
#pragma unroll
      for (int e = 0; e < 8; ++e) {
        v2h t = {q0[e], q1[e]};
        *(v2h*)&sBt[(col8 + e) * 64 + k] = t;
      }
    }
    __syncthreads();
#pragma unroll
    for (int kk = 0; kk < 64; kk += 32) {
      const v16h a0 = frag_a(sA + (wrow * 32) * 64 + kk, 64, lane);
      const v16h a1 = frag_a(sA + (wrow * 32 + 16) * 64 + kk, 64, lane);
#pragma unroll
      for (int j = 0; j < 4; ++j) {
        const v16h bf = frag_bt(sBt + (wcol * 64 + j * 16) * 64 + kk, 64, lane);
        acc[j]     = WMMA_F16(a0, bf, acc[j]);
        acc[4 + j] = WMMA_F16(a1, bf, acc[4 + j]);
      }
    }
  }

  const int lc = lane & 15, lr = (lane >> 4) << 3;
#pragma unroll
  for (int ri = 0; ri < 2; ++ri) {
#pragma unroll
    for (int j = 0; j < 4; ++j) {
      const v8f c = acc[ri * 4 + j];
      const int cc = m0 + wcol * 64 + j * 16 + lc;
#pragma unroll
      for (int el = 0; el < 8; ++el) {
        const int rr = n0 + wrow * 32 + ri * 16 + lr + el;
        Eng[(size_t)rr * 4096 + cc] = c[el];
      }
    }
  }
}

// ---------------------------------------------------------------------------
// Kernel 4: per-row stats of e: Mrow[n] = min_m e[n][m],
//           RZ[n] = 1 / sum_m exp(Mrow[n] - e[n][m]).
// (Reference's softmax(rowmax - e) == exp(rowmin - e)/Z.)
// ---------------------------------------------------------------------------
__global__ __launch_bounds__(256) void pam_stats(const float* __restrict__ Eng,
                                                 float* __restrict__ Mrow,
                                                 float* __restrict__ RZ) {
  __shared__ float red[256];
  const int t = threadIdx.x;
  const size_t n = blockIdx.x;
  const float* src = Eng + n * 4096;
  float r[16];
  float mn = 3.4e38f;
#pragma unroll
  for (int i = 0; i < 16; ++i) { r[i] = src[t + i * 256]; mn = fminf(mn, r[i]); }
  red[t] = mn; __syncthreads();
  for (int s = 128; s > 0; s >>= 1) { if (t < s) red[t] = fminf(red[t], red[t + s]); __syncthreads(); }
  mn = red[0]; __syncthreads();
  float sm = 0.f;
#pragma unroll
  for (int i = 0; i < 16; ++i) sm += __expf(mn - r[i]);
  red[t] = sm; __syncthreads();
  for (int s = 128; s > 0; s >>= 1) { if (t < s) red[t] += red[t + s]; __syncthreads(); }
  if (t == 0) { Mrow[n] = mn; RZ[n] = 1.0f / red[0]; }
}

// ---------------------------------------------------------------------------
// Kernel 5: out[c][m] = sum_n Vh[c][n] * (exp(Mrow[n]-e[n][m]) * RZ[n]).
// attn tile materialized on the fly (f32 energy -> f16 LDS, transposed).
// grid = (32 mTiles, 2 cTiles), block 256; K = 4096 in chunks of 32.
// ---------------------------------------------------------------------------
__global__ __launch_bounds__(256) void pam_out(const _Float16* __restrict__ Vh,
                                               const float* __restrict__ Eng,
                                               const float* __restrict__ Mrow,
                                               const float* __restrict__ RZ,
                                               float* __restrict__ Outp, int b) {
  __shared__ __align__(16) _Float16 sA[128 * 32];   // [c-row][K]  K-chunk = 32
  __shared__ __align__(16) _Float16 sBt[128 * 32];  // [m-col][K]
  const int tid = threadIdx.x, lane = tid & 31, wave = tid >> 5;
  const int wrow = wave & 3, wcol = wave >> 2;
  const int m0 = blockIdx.x * 128, c0 = blockIdx.y * 128;

  v8f acc[8];
#pragma unroll
  for (int i = 0; i < 8; ++i) acc[i] = zero8();

  for (int kn0 = 0; kn0 < 4096; kn0 += 32) {
    __syncthreads();
    // A panel: Vh straight copy (2 x 16B per thread) -- async-to-LDS if available.
#if defined(HAVE_ASYNC_LDS)
#pragma unroll
    for (int it = 0; it < 2; ++it) {
      const int i = tid + it * 256;
      const int r = i >> 2, c8 = (i & 3) << 3;
      const _Float16* gsrc = Vh + (size_t)(b * 256 + c0 + r) * 4096 + kn0 + c8;
      __builtin_amdgcn_global_load_async_to_lds_b128(
          (gbl_v4i*)gsrc, (lds_v4i*)&sA[r * 32 + c8], 0, 0);
    }
#else
#pragma unroll
    for (int it = 0; it < 2; ++it) {
      const int i = tid + it * 256;
      const int r = i >> 2, c8 = (i & 3) << 3;
      *(v8h*)&sA[r * 32 + c8] =
          *(const v8h*)&Vh[(size_t)(b * 256 + c0 + r) * 4096 + kn0 + c8];
    }
#endif
    // B panel: f32 energy -> exp(M-e)*RZ -> f16, transposed into [col][32].
    {
      const int k2 = tid >> 4, col8 = (tid & 15) << 3;
      const int k = k2 << 1;
      const float* s0 = Eng + (size_t)(kn0 + k) * 4096 + m0 + col8;
      fl4 r0a = *(const fl4*)(s0);
      fl4 r0b = *(const fl4*)(s0 + 4);
      fl4 r1a = *(const fl4*)(s0 + 4096);
      fl4 r1b = *(const fl4*)(s0 + 4100);
      const float m0f = Mrow[kn0 + k], z0 = RZ[kn0 + k];
      const float m1f = Mrow[kn0 + k + 1], z1 = RZ[kn0 + k + 1];
#pragma unroll
      for (int e = 0; e < 4; ++e) {
        v2h t0 = {(_Float16)(__expf(m0f - r0a[e]) * z0), (_Float16)(__expf(m1f - r1a[e]) * z1)};
        v2h t1 = {(_Float16)(__expf(m0f - r0b[e]) * z0), (_Float16)(__expf(m1f - r1b[e]) * z1)};
        *(v2h*)&sBt[(col8 + e) * 32 + k]     = t0;
        *(v2h*)&sBt[(col8 + 4 + e) * 32 + k] = t1;
      }
    }
    if (kn0 + 32 < 4096) {  // global_prefetch_b8 on the streamed energy matrix
      __builtin_prefetch(&Eng[(size_t)(kn0 + 32 + (tid >> 7)) * 4096 + m0 + (tid & 127)], 0, 1);
    }
#if defined(HAVE_ASYNC_LDS)
    __builtin_amdgcn_s_wait_asynccnt(0);
#endif
    __syncthreads();
    const v16h a0 = frag_a(sA + (wrow * 32) * 32, 32, lane);
    const v16h a1 = frag_a(sA + (wrow * 32 + 16) * 32, 32, lane);
#pragma unroll
    for (int j = 0; j < 4; ++j) {
      const v16h bf = frag_bt(sBt + (wcol * 64 + j * 16) * 32, 32, lane);
      acc[j]     = WMMA_F16(a0, bf, acc[j]);
      acc[4 + j] = WMMA_F16(a1, bf, acc[4 + j]);
    }
  }

  const int lc = lane & 15, lr = (lane >> 4) << 3;
#pragma unroll
  for (int ri = 0; ri < 2; ++ri) {
#pragma unroll
    for (int j = 0; j < 4; ++j) {
      const v8f c = acc[ri * 4 + j];
      const int cc = m0 + wcol * 64 + j * 16 + lc;
#pragma unroll
      for (int el = 0; el < 8; ++el) {
        const int rr = c0 + wrow * 32 + ri * 16 + lr + el;
        Outp[(size_t)rr * 4096 + cc] = c[el];
      }
    }
  }
}

// ---------------------------------------------------------------------------
// Kernel 6: final row softmax + residual: y = x + gamma * softmax_m(out[c][:]).
// ---------------------------------------------------------------------------
__global__ __launch_bounds__(256) void pam_final(const float* __restrict__ Outp,
                                                 const float* __restrict__ x,
                                                 const float* __restrict__ gamma,
                                                 float* __restrict__ out, int b) {
  __shared__ float red[256];
  const int t = threadIdx.x;
  const int c = blockIdx.x;
  const float* src = Outp + (size_t)c * 4096;
  float r[16];
  float mx = -3.4e38f;
#pragma unroll
  for (int i = 0; i < 16; ++i) { r[i] = src[t + i * 256]; mx = fmaxf(mx, r[i]); }
  red[t] = mx; __syncthreads();
  for (int s = 128; s > 0; s >>= 1) { if (t < s) red[t] = fmaxf(red[t], red[t + s]); __syncthreads(); }
  mx = red[0]; __syncthreads();
  float sm = 0.f;
#pragma unroll
  for (int i = 0; i < 16; ++i) { r[i] = __expf(r[i] - mx); sm += r[i]; }
  red[t] = sm; __syncthreads();
  for (int s = 128; s > 0; s >>= 1) { if (t < s) red[t] += red[t + s]; __syncthreads(); }
  const float rinv = 1.0f / red[0];
  const float g = gamma[0];
  const size_t base = (size_t)(b * 256 + c) * 4096;
#pragma unroll
  for (int i = 0; i < 16; ++i) {
    const int idx = t + i * 256;
    out[base + idx] = x[base + idx] + g * (r[i] * rinv);
  }
}

// ---------------------------------------------------------------------------
extern "C" void kernel_launch(void* const* d_in, const int* in_sizes, int n_in,
                              void* d_out, int out_size, void* d_ws, size_t ws_size,
                              hipStream_t stream) {
  (void)in_sizes; (void)n_in; (void)out_size; (void)ws_size;
  const float* x     = (const float*)d_in[0];
  const float* Wq    = (const float*)d_in[1];
  const float* bq    = (const float*)d_in[2];
  const float* Wk    = (const float*)d_in[3];
  const float* bk    = (const float*)d_in[4];
  const float* Wv    = (const float*)d_in[5];
  const float* bv    = (const float*)d_in[6];
  const float* gamma = (const float*)d_in[7];
  float* out = (float*)d_out;

  char* ws = (char*)d_ws;
  _Float16* Qt   = (_Float16*)(ws);                                  //  4 MB  [B][128][N]
  _Float16* Kn   = (_Float16*)(ws + (size_t)(4)  * (1u << 20));      //  4 MB  [B][N][128]
  float*    Vraw = (float*)   (ws + (size_t)(8)  * (1u << 20));      // 16 MB  [B][C][N]
  _Float16* Vh   = (_Float16*)(ws + (size_t)(24) * (1u << 20));      //  8 MB  [B][C][N]
  float*    Mrow = (float*)   (ws + (size_t)(32) * (1u << 20));      // 16 KB  [N]   (per batch)
  float*    RZ   = (float*)   (ws + (size_t)(32) * (1u << 20) + (64u << 10)); // 16 KB
  float*    Eng  = (float*)   (ws + (size_t)(33) * (1u << 20));      // 64 MB  [N][N] (per batch)
  float*    Outp = (float*)   (ws + (size_t)(97) * (1u << 20));      //  4 MB  [C][N] (per batch)

  pam_proj<<<dim3(128, 4), 256, 0, stream>>>(x, Wq, bq, Wk, bk, Wv, bv, Qt, Kn, Vraw);
  pam_vsoftmax<<<1024, 256, 0, stream>>>(Vraw, Vh);
  for (int b = 0; b < 4; ++b) {
    pam_energy<<<dim3(32, 32), 256, 0, stream>>>(Kn, Qt, Eng, b);
    pam_stats<<<4096, 256, 0, stream>>>(Eng, Mrow, RZ);
    pam_out<<<dim3(32, 2), 256, 0, stream>>>(Vh, Eng, Mrow, RZ, Outp, b);
    pam_final<<<256, 256, 0, stream>>>(Outp, x, gamma, out, b);
  }
}
